// HQQGroupedGemm_55430847922806
// MI455X (gfx1250) — matrix operand order repaired and checked by reading, not compiled, not run on hardware
//
#include <hip/hip_runtime.h>

typedef __attribute__((ext_vector_type(16))) _Float16 v16h;
typedef __attribute__((ext_vector_type(8)))  _Float16 v8h;
typedef __attribute__((ext_vector_type(2)))  _Float16 h2;
typedef __attribute__((ext_vector_type(8)))  float    v8f;
typedef __attribute__((ext_vector_type(4)))  float    v4f;
typedef __attribute__((ext_vector_type(4)))  int      v4i;

#define E_   8
#define IN_  1024
#define OUT_ 2816
#define GS_  64
#define G_   16
#define N_   2048

#define BM  128
#define BN  128
#define BK  32
#define PAD 8
#define LDK (BK + PAD)   // 40 halves = 80 B/row: 16B-aligned rows, conflict-free b128 frag loads

#define NSTEP (IN_ / BK) // 32 K-steps

__global__ __launch_bounds__(256)
void hqq_grouped_gemm_wmma(const float* __restrict__ x,
                           const int*   __restrict__ tpe,
                           const int*   __restrict__ Wq,
                           const float* __restrict__ scales,
                           const float* __restrict__ zeros,
                           float*       __restrict__ y)
{
    // double-buffered tiles
    __shared__ _Float16 As[2][BM][LDK];   // x tile, row-major [m][k], f16
    __shared__ _Float16 Bs[2][BN][LDK];   // dequantized W tile, n-major [n][k], f16

    const int e  = blockIdx.z;
    const int mt = blockIdx.y;
    const int n0 = blockIdx.x * BN;

    // exclusive prefix sum over the 8 expert counts (tokens contiguous per expert)
    int start = 0;
    #pragma unroll
    for (int i = 0; i < E_; ++i) {
        int c = tpe[i];
        if (i < e) start += c;
    }
    const int count = tpe[e];
    if (mt * BM >= count) return;               // over-provisioned tile: nothing to do
    const int row0       = start + mt * BM;
    const int rows_valid = min(BM, count - mt * BM);

    const int t    = threadIdx.x;
    const int lane = t & 31;
    const int wave = t >> 5;
    const int wm   = wave >> 2;     // 0..1  -> 64 rows
    const int wn   = wave & 3;      // 0..3  -> 32 cols
    const int l16  = lane & 15;
    const int lh   = lane >> 4;

    // ---- staging coordinates ----
    // x tile: thread -> (row, 16-wide k half)
    const int xr = t >> 1;          // 0..127
    const int xh = (t & 1) * 16;    // 0 / 16
    const bool xvalid = xr < rows_valid;
    // W tile: thread -> 4(k) x 4(n) micro-tile
    const int qn = (t & 31) * 4;    // n offset within tile, 4 wide
    const int qk = (t >> 5) * 4;    // k offset within tile, 4 rows

    // hoisted global base pointers
    const float* xp    = x + (size_t)(row0 + xr) * IN_ + xh;              // deref only if xvalid
    const int*   wbase = Wq + ((size_t)e * IN_ + qk) * OUT_ + n0 + qn;
    const float* sbase = scales + (size_t)e * G_ * OUT_ + n0 + qn;
    const float* zbase = zeros  + (size_t)e * G_ * OUT_ + n0 + qn;

    // ---- staging registers (live across the WMMA block for pipelining) ----
    v4f xv0, xv1, xv2, xv3;
    v4i q0, q1, q2, q3;             // 4 k-rows of 4 codes
    v4f sv, zv;                     // per-n scale / zero (group fixed within a K-step)

    auto load_stage = [&](int k0) {
        if (xvalid) {
            const float* p = xp + k0;
            xv0 = *(const v4f*)(p + 0);
            xv1 = *(const v4f*)(p + 4);
            xv2 = *(const v4f*)(p + 8);
            xv3 = *(const v4f*)(p + 12);
        } else {
            v4f zz = {0.f, 0.f, 0.f, 0.f};
            xv0 = xv1 = xv2 = xv3 = zz;
        }
        const int* wpk = wbase + (size_t)k0 * OUT_;
        q0 = *(const v4i*)(wpk + 0 * OUT_);
        q1 = *(const v4i*)(wpk + 1 * OUT_);
        q2 = *(const v4i*)(wpk + 2 * OUT_);
        q3 = *(const v4i*)(wpk + 3 * OUT_);
        const int g = k0 >> 6;                       // GS=64: group constant per K-step
        sv = *(const v4f*)(sbase + (size_t)g * OUT_);
        zv = *(const v4f*)(zbase + (size_t)g * OUT_);
        if (k0 + 2 * BK < IN_)                       // far prefetch on the dominant stream
            __builtin_prefetch(wpk + (size_t)(2 * BK) * OUT_, 0, 1);
    };

    auto store_stage = [&](int b) {
        // x: f32 -> f16, two b128 stores
        v8h lo, hi;
        #pragma unroll
        for (int c = 0; c < 4; ++c) {
            lo[c]     = (_Float16)xv0[c];
            lo[c + 4] = (_Float16)xv1[c];
            hi[c]     = (_Float16)xv2[c];
            hi[c + 4] = (_Float16)xv3[c];
        }
        *(v8h*)&As[b][xr][xh]     = lo;
        *(v8h*)&As[b][xr][xh + 8] = hi;
        // W: dequant + transpose to n-major; packed h2 (b32) stores, 2-way conflicts max
        #pragma unroll
        for (int j = 0; j < 4; ++j) {
            const float s = sv[j], z = zv[j];
            h2 p0, p1;
            p0[0] = (_Float16)(((float)q0[j] - z) * s);
            p0[1] = (_Float16)(((float)q1[j] - z) * s);
            p1[0] = (_Float16)(((float)q2[j] - z) * s);
            p1[1] = (_Float16)(((float)q3[j] - z) * s);
            *(h2*)&Bs[b][qn + j][qk]     = p0;
            *(h2*)&Bs[b][qn + j][qk + 2] = p1;
        }
    };

    v8f acc[4][2];
    #pragma unroll
    for (int i = 0; i < 4; ++i)
        #pragma unroll
        for (int j = 0; j < 2; ++j) {
            v8f zz = {0.f, 0.f, 0.f, 0.f, 0.f, 0.f, 0.f, 0.f};
            acc[i][j] = zz;
        }

    auto compute = [&](int b) {
        // fragment layouts: M/N per ISA 7.12.2; k order is this code's own contiguous one
        const int ka = lh * 16;     // A 16x32, contiguous k: lanes 0-15 K=0..15; lanes 16-31 K=16..31
        const int kb = lh * 16;     // B 32x16, same contiguous k: lanes 0-15 K=0..15; lanes 16-31 K=16..31
        v16h afrag[4];
        #pragma unroll
        for (int i = 0; i < 4; ++i) {
            const int m = wm * 64 + i * 16 + l16;
            v8h alo = *(const v8h*)&As[b][m][ka];
            v8h ahi = *(const v8h*)&As[b][m][ka + 8];
            #pragma unroll
            for (int c = 0; c < 8; ++c) { afrag[i][c] = alo[c]; afrag[i][c + 8] = ahi[c]; }
        }
        v16h bfrag[2];
        #pragma unroll
        for (int j = 0; j < 2; ++j) {
            const int n = wn * 32 + j * 16 + l16;
            v8h blo = *(const v8h*)&Bs[b][n][kb];
            v8h bhi = *(const v8h*)&Bs[b][n][kb + 8];
            #pragma unroll
            for (int c = 0; c < 8; ++c) { bfrag[j][c] = blo[c]; bfrag[j][c + 8] = bhi[c]; }
        }
        #pragma unroll
        for (int i = 0; i < 4; ++i)
            #pragma unroll
            for (int j = 0; j < 2; ++j)
                acc[i][j] = __builtin_amdgcn_wmma_f32_16x16x32_f16(
                    false, afrag[i], false, bfrag[j],
                    (short)0, acc[i][j], false, false);
    };

    // ---- software pipeline: load(s) || compute(s-1) ----
    load_stage(0);
    store_stage(0);
    __syncthreads();

    int buf = 0;
    for (int s = 1; s < NSTEP; ++s) {
        load_stage(s * BK);         // global loads for step s go in flight
        compute(buf);               // WMMAs on step s-1 hide their latency
        store_stage(buf ^ 1);       // wait loads, dequant/convert, fill other buffer
        __syncthreads();
        buf ^= 1;
    }
    compute(buf);

    // ---- epilogue: C/D layout: N = lane%16, M = r + 8*(lane>=16) ----
    #pragma unroll
    for (int i = 0; i < 4; ++i) {
        #pragma unroll
        for (int j = 0; j < 2; ++j) {
            const int col = n0 + wn * 32 + j * 16 + l16;
            #pragma unroll
            for (int r = 0; r < 8; ++r) {
                const int row = wm * 64 + i * 16 + lh * 8 + r;
                if (row < rows_valid)
                    y[(size_t)(row0 + row) * OUT_ + col] = acc[i][j][r];
            }
        }
    }
}

extern "C" void kernel_launch(void* const* d_in, const int* in_sizes, int n_in,
                              void* d_out, int out_size, void* d_ws, size_t ws_size,
                              hipStream_t stream) {
    const float* x      = (const float*)d_in[0];
    const int*   tpe    = (const int*)  d_in[1];
    const int*   Wq     = (const int*)  d_in[2];
    const float* scales = (const float*)d_in[3];
    const float* zeros  = (const float*)d_in[4];
    float*       y      = (float*)d_out;

    dim3 grid(OUT_ / BN, N_ / BM, E_);   // (22, 16, 8); empty expert tiles early-exit
    dim3 block(256);                     // 8 wave32s
    hqq_grouped_gemm_wmma<<<grid, block, 0, stream>>>(x, tpe, Wq, scales, zeros, y);
}
